// EquivariantGraphConvolutionalLayer_27994596835771
// MI455X (gfx1250) — matrix operand order repaired and checked by reading, not compiled, not run on hardware
//
#include <hip/hip_runtime.h>
#include <hip/hip_bf16.h>

#define NNODES 10000
#define NEDGES 160000
#define NDIMS  3
#define INDIM  128
#define HID    128
#define ROWS_E (NEDGES * NDIMS)   /* 480000 -> 15000 wave-pairs of 32 rows */
#define ROWS_N (NNODES * NDIMS)   /* 30000  -> 1875 tiles of 16           */
#define K1     288                /* 2*INDIM+1 = 257 padded to 288 */
#define K1S    296                /* LDS row stride (halves) for msg_in */
#define KN     256                /* node concat K */
#define KNS    264                /* LDS row stride for node concat */
#define HS     136                /* LDS row stride for 128-wide hidden */

typedef __attribute__((ext_vector_type(16))) _Float16 v16h;
typedef __attribute__((ext_vector_type(8)))  _Float16 v8h;
typedef __attribute__((ext_vector_type(4)))  _Float16 v4h;
typedef __attribute__((ext_vector_type(8)))  float    v8f;

__device__ __forceinline__ float swishf(float x) {
    return x / (1.0f + __expf(-x));
}

// A-fragment (16x32 f16) from LDS, row-major with given stride (halves).
__device__ __forceinline__ v16h load_a_frag_lds(const _Float16* base, int stride, int k0, int lane) {
    int row = lane & 15;
    int kb  = (lane >> 4) << 3;      // 0 or 8
    const _Float16* p = base + row * stride + k0 + kb;
    union { v16h v; v8h h[2]; } u;
    u.h[0] = *(const v8h*)(p);
    u.h[1] = *(const v8h*)(p + 16);
    return u.v;
}

// B-fragment (32x16 f16) from transposed weights Wt[N][Kp] in global memory.
__device__ __forceinline__ v16h load_b_frag(const _Float16* Wt, int Kp, int ntile, int k0, int lane) {
    int n    = lane & 15;
    int koff = (lane >> 4) << 3;     // 0 or 8
    const _Float16* p = Wt + (size_t)(ntile * 16 + n) * Kp + k0 + koff;
    union { v16h v; v8h h[2]; } u;
    u.h[0] = *(const v8h*)(p);
    u.h[1] = *(const v8h*)(p + 16);
    return u.v;
}

__device__ __forceinline__ v8f wmma16(v16h a, v16h b, v8f c) {
    return __builtin_amdgcn_wmma_f32_16x16x32_f16(false, a, false, b, (short)0, c, false, false);
}

__device__ __forceinline__ void cvt_store4(_Float16* dst, float4 v) {
    v4h h;
    h[0] = (_Float16)v.x; h[1] = (_Float16)v.y;
    h[2] = (_Float16)v.z; h[3] = (_Float16)v.w;
    *(v4h*)dst = h;
}

// ---------------------------------------------------------------------------
// Weight prep: f32 -> f16, transposed to [N][K], We1 K padded 257 -> 288.
// ---------------------------------------------------------------------------
__global__ void egnn_prep_kernel(const float* __restrict__ We1, const float* __restrict__ We2,
                                 const float* __restrict__ Wc1, const float* __restrict__ Wn1,
                                 const float* __restrict__ Wn2, const float* __restrict__ Wv1,
                                 _Float16* __restrict__ We1t, _Float16* __restrict__ We2t,
                                 _Float16* __restrict__ Wc1t, _Float16* __restrict__ Wn1t,
                                 _Float16* __restrict__ Wn2t, _Float16* __restrict__ Wv1t) {
    int idx = blockIdx.x * 256 + threadIdx.x;
    if (idx < 128 * K1) {
        int n = idx / K1, k = idx % K1;
        We1t[idx] = (_Float16)((k < 2 * INDIM + 1) ? We1[k * 128 + n] : 0.0f);
    }
    if (idx < 128 * 256) {
        int n = idx / 256, k = idx % 256;
        Wn1t[idx] = (_Float16)Wn1[k * 128 + n];
    }
    if (idx < 128 * 128) {
        int n = idx / 128, k = idx % 128;
        We2t[idx] = (_Float16)We2[k * 128 + n];
        Wc1t[idx] = (_Float16)Wc1[k * 128 + n];
        Wn2t[idx] = (_Float16)Wn2[k * 128 + n];
        Wv1t[idx] = (_Float16)Wv1[k * 128 + n];
    }
}

// ---------------------------------------------------------------------------
// Edge kernel: each wave owns TWO 16-row M-tiles (32 rows of (edge,dim)).
// Every B fragment is loaded once and used by two WMMAs -> 2x B reuse.
//   m1 = swish(msg_in @ We1 + be1)      (9 K x 8 N x 2 tiles WMMA)
//   m  = swish(m1 @ We2 + be2)          (4 x 8 x 2)  -> atomicAdd agg
//   c  = swish(m @ Wc1 + bc1) @ Wc2+bc2 (4 x 8 x 2)  -> atomicAdd upd_coord
// ---------------------------------------------------------------------------
__global__ __launch_bounds__(64) void egnn_edge_kernel(
    const float* __restrict__ node_feat, const float* __restrict__ coordv,
    const long long* __restrict__ ei,
    const _Float16* __restrict__ We1t, const float* __restrict__ be1,
    const _Float16* __restrict__ We2t, const float* __restrict__ be2,
    const _Float16* __restrict__ Wc1t, const float* __restrict__ bc1,
    const float* __restrict__ Wc2, const float* __restrict__ bc2,
    float* __restrict__ agg, float* __restrict__ upd_coord) {
    __shared__ _Float16 sA[2][32 * K1S];
    __shared__ _Float16 sM[2][32 * HS];
    __shared__ float    sRed[2][512];

    int wave = threadIdx.x >> 5;
    int lane = threadIdx.x & 31;
    int pairIdx = blockIdx.x * 2 + wave;   // grid = 7500 -> exactly 15000 pairs
    int row0 = pairIdx * 32;

    _Float16* A   = &sA[wave][0];
    _Float16* M1  = &sM[wave][0];
    _Float16* M2  = &sA[wave][0];          // reuse msg_in region after layer1
    float*    RED = &sRed[wave][0];

    // ---- build 32-row msg_in tile in LDS (f16, padded to K1); 1 lane/row ----
    {
        int row = lane;
        int rg  = row0 + row;
        int edge = rg / 3, dim = rg % 3;
        int src = (int)ei[edge];
        int tgt = (int)ei[NEDGES + edge];
        const float* ft = node_feat + ((size_t)tgt * NDIMS + dim) * INDIM;
        const float* fs = node_feat + ((size_t)src * NDIMS + dim) * INDIM;
        _Float16* dst = A + row * K1S;
#pragma unroll
        for (int k = 0; k < INDIM; k += 4) {
            cvt_store4(dst + k,         *(const float4*)(ft + k));
            cvt_store4(dst + INDIM + k, *(const float4*)(fs + k));
        }
        float dx = coordv[tgt * 3 + 0] - coordv[src * 3 + 0];
        float dy = coordv[tgt * 3 + 1] - coordv[src * 3 + 1];
        float dz = coordv[tgt * 3 + 2] - coordv[src * 3 + 2];
        float dist = dx * dx + dy * dy + dz * dz;
        dst[2 * INDIM] = (_Float16)dist;
#pragma unroll
        for (int k = 2 * INDIM + 1; k < K1; k++) dst[k] = (_Float16)0.0f;
    }
    __syncthreads();

    int col_l = lane & 15;
    int rbase = (lane >> 4) << 3;

    // ---- layer 1: m1 = swish(msg_in @ We1 + be1) ----
    v16h a1[2][9];
#pragma unroll
    for (int t = 0; t < 2; t++)
#pragma unroll
        for (int kk = 0; kk < 9; kk++)
            a1[t][kk] = load_a_frag_lds(A + t * 16 * K1S, K1S, kk * 32, lane);

    for (int nt = 0; nt < 8; nt++) {
        float b = be1[nt * 16 + col_l];
        v8f acc0, acc1;
#pragma unroll
        for (int i = 0; i < 8; i++) { acc0[i] = b; acc1[i] = b; }
#pragma unroll
        for (int kk = 0; kk < 9; kk++) {
            v16h bb = load_b_frag(We1t, K1, nt, kk * 32, lane);
            acc0 = wmma16(a1[0][kk], bb, acc0);
            acc1 = wmma16(a1[1][kk], bb, acc1);
        }
#pragma unroll
        for (int i = 0; i < 8; i++) {
            M1[(rbase + i) * HS + nt * 16 + col_l]        = (_Float16)swishf(acc0[i]);
            M1[(16 + rbase + i) * HS + nt * 16 + col_l]   = (_Float16)swishf(acc1[i]);
        }
    }
    __syncthreads();

    // ---- layer 2: m = swish(m1 @ We2 + be2); scatter-add into agg ----
    v16h a2[2][4];
#pragma unroll
    for (int t = 0; t < 2; t++)
#pragma unroll
        for (int kk = 0; kk < 4; kk++)
            a2[t][kk] = load_a_frag_lds(M1 + t * 16 * HS, HS, kk * 32, lane);

    size_t aggRow[2][8];
#pragma unroll
    for (int t = 0; t < 2; t++)
#pragma unroll
        for (int i = 0; i < 8; i++) {
            int rg = row0 + t * 16 + rbase + i;
            int edge = rg / 3, dim = rg % 3;
            int tgt = (int)ei[NEDGES + edge];
            aggRow[t][i] = ((size_t)tgt * 3 + dim) * (size_t)HID;
        }

    for (int nt = 0; nt < 8; nt++) {
        float b = be2[nt * 16 + col_l];
        v8f acc0, acc1;
#pragma unroll
        for (int i = 0; i < 8; i++) { acc0[i] = b; acc1[i] = b; }
#pragma unroll
        for (int kk = 0; kk < 4; kk++) {
            v16h bb = load_b_frag(We2t, 128, nt, kk * 32, lane);
            acc0 = wmma16(a2[0][kk], bb, acc0);
            acc1 = wmma16(a2[1][kk], bb, acc1);
        }
#pragma unroll
        for (int i = 0; i < 8; i++) {
            float m0 = swishf(acc0[i]);
            float m1v = swishf(acc1[i]);
            atomicAdd(&agg[aggRow[0][i] + nt * 16 + col_l], m0);
            atomicAdd(&agg[aggRow[1][i] + nt * 16 + col_l], m1v);
            M2[(rbase + i) * HS + nt * 16 + col_l]      = (_Float16)m0;
            M2[(16 + rbase + i) * HS + nt * 16 + col_l] = (_Float16)m1v;
        }
    }
    __syncthreads();

    // ---- layer 3: c = swish(m @ Wc1 + bc1) @ Wc2 + bc2 ----
    v16h a3[2][4];
#pragma unroll
    for (int t = 0; t < 2; t++)
#pragma unroll
        for (int kk = 0; kk < 4; kk++)
            a3[t][kk] = load_a_frag_lds(M2 + t * 16 * HS, HS, kk * 32, lane);

    float pacc[2][8];
#pragma unroll
    for (int t = 0; t < 2; t++)
#pragma unroll
        for (int i = 0; i < 8; i++) pacc[t][i] = 0.0f;

    for (int nt = 0; nt < 8; nt++) {
        float b = bc1[nt * 16 + col_l];
        v8f acc0, acc1;
#pragma unroll
        for (int i = 0; i < 8; i++) { acc0[i] = b; acc1[i] = b; }
#pragma unroll
        for (int kk = 0; kk < 4; kk++) {
            v16h bb = load_b_frag(Wc1t, 128, nt, kk * 32, lane);
            acc0 = wmma16(a3[0][kk], bb, acc0);
            acc1 = wmma16(a3[1][kk], bb, acc1);
        }
        float w2 = Wc2[nt * 16 + col_l];
#pragma unroll
        for (int i = 0; i < 8; i++) {
            pacc[0][i] += swishf(acc0[i]) * w2;
            pacc[1][i] += swishf(acc1[i]) * w2;
        }
    }
#pragma unroll
    for (int t = 0; t < 2; t++)
#pragma unroll
        for (int i = 0; i < 8; i++)
            RED[(t * 16 + rbase + i) * 16 + col_l] = pacc[t][i];
    __syncthreads();

    {   // 32 rows reduced by 32 lanes
        float c = bc2[0];
#pragma unroll
        for (int j = 0; j < 16; j++) c += RED[lane * 16 + j];
        int rg = row0 + lane;
        int edge = rg / 3, dim = rg % 3;
        int src = (int)ei[edge];
        int tgt = (int)ei[NEDGES + edge];
        float rel = coordv[tgt * 3 + dim] - coordv[src * 3 + dim];
        atomicAdd(&upd_coord[tgt * 3 + dim], rel * c);
    }
}

// ---------------------------------------------------------------------------
// Node kernel: node_out = node_feat + swish([nf,agg]@Wn1+bn1)@Wn2+bn2
//              v = swish(nf@Wv1+bv1)@Wv2+bv2 ; vel = velocity * v
//              coord = coordinate + upd_coord/degree + vel
// ---------------------------------------------------------------------------
__global__ __launch_bounds__(128) void egnn_node_kernel(
    const float* __restrict__ node_feat, const float* __restrict__ degree,
    const float* __restrict__ coordv, const float* __restrict__ velocity,
    const _Float16* __restrict__ Wn1t, const float* __restrict__ bn1,
    const _Float16* __restrict__ Wn2t, const float* __restrict__ bn2,
    const _Float16* __restrict__ Wv1t, const float* __restrict__ bv1,
    const float* __restrict__ Wv2, const float* __restrict__ bv2,
    const float* __restrict__ agg, const float* __restrict__ upd_coord,
    float* __restrict__ out_coord, float* __restrict__ out_node,
    float* __restrict__ out_vel) {
    __shared__ _Float16 sA[4][16 * KNS];
    __shared__ _Float16 sH[4][16 * HS];
    __shared__ float    sRed[4][256];

    int wave = threadIdx.x >> 5;
    int lane = threadIdx.x & 31;
    int tile = blockIdx.x * 4 + wave;
    int row0 = tile * 16;

    _Float16* A   = &sA[wave][0];
    _Float16* H   = &sH[wave][0];
    float*    RED = &sRed[wave][0];

    // ---- build [node_feat | agg] tile in LDS ----
    {
        int row  = lane >> 1;
        int part = lane & 1;
        int rg   = row0 + row;
        int rgc  = (rg < ROWS_N) ? rg : 0;
        const float* f = part ? (agg + (size_t)rgc * HID)
                              : (node_feat + (size_t)rgc * INDIM);
        _Float16* dst = A + row * KNS + part * 128;
#pragma unroll
        for (int k = 0; k < 128; k += 4)
            cvt_store4(dst + k, *(const float4*)(f + k));
    }
    __syncthreads();

    int col_l = lane & 15;
    int rbase = (lane >> 4) << 3;

    // ---- h1 = swish([nf,agg] @ Wn1 + bn1) ----
    v16h an[8];
#pragma unroll
    for (int kk = 0; kk < 8; kk++) an[kk] = load_a_frag_lds(A, KNS, kk * 32, lane);

    for (int nt = 0; nt < 8; nt++) {
        float b = bn1[nt * 16 + col_l];
        v8f acc;
#pragma unroll
        for (int i = 0; i < 8; i++) acc[i] = b;
#pragma unroll
        for (int kk = 0; kk < 8; kk++)
            acc = wmma16(an[kk], load_b_frag(Wn1t, 256, nt, kk * 32, lane), acc);
#pragma unroll
        for (int i = 0; i < 8; i++)
            H[(rbase + i) * HS + nt * 16 + col_l] = (_Float16)swishf(acc[i]);
    }
    __syncthreads();

    // ---- node_out = node_feat + h1 @ Wn2 + bn2 ----
    v16h ah[4];
#pragma unroll
    for (int kk = 0; kk < 4; kk++) ah[kk] = load_a_frag_lds(H, HS, kk * 32, lane);

    for (int nt = 0; nt < 8; nt++) {
        float b = bn2[nt * 16 + col_l];
        v8f acc;
#pragma unroll
        for (int i = 0; i < 8; i++) acc[i] = b;
#pragma unroll
        for (int kk = 0; kk < 4; kk++)
            acc = wmma16(ah[kk], load_b_frag(Wn2t, 128, nt, kk * 32, lane), acc);
#pragma unroll
        for (int i = 0; i < 8; i++) {
            int rg = row0 + rbase + i;
            if (rg < ROWS_N) {
                size_t idx = (size_t)rg * INDIM + nt * 16 + col_l;
                out_node[idx] = node_feat[idx] + acc[i];
            }
        }
    }

    // ---- velocity branch: vh = swish(nf @ Wv1 + bv1); v = vh . Wv2 + bv2 ----
    v16h av[4];
#pragma unroll
    for (int kk = 0; kk < 4; kk++) av[kk] = load_a_frag_lds(A, KNS, kk * 32, lane);

    float pacc[8];
#pragma unroll
    for (int i = 0; i < 8; i++) pacc[i] = 0.0f;

    for (int nt = 0; nt < 8; nt++) {
        float b = bv1[nt * 16 + col_l];
        v8f acc;
#pragma unroll
        for (int i = 0; i < 8; i++) acc[i] = b;
#pragma unroll
        for (int kk = 0; kk < 4; kk++)
            acc = wmma16(av[kk], load_b_frag(Wv1t, 128, nt, kk * 32, lane), acc);
        float w2 = Wv2[nt * 16 + col_l];
#pragma unroll
        for (int i = 0; i < 8; i++) pacc[i] += swishf(acc[i]) * w2;
    }
#pragma unroll
    for (int i = 0; i < 8; i++) RED[(rbase + i) * 16 + col_l] = pacc[i];
    __syncthreads();

    if (lane < 16) {
        int rg = row0 + lane;
        if (rg < ROWS_N) {
            float v = bv2[0];
#pragma unroll
            for (int j = 0; j < 16; j++) v += RED[lane * 16 + j];
            int node = rg / 3;
            float vel = velocity[rg] * v;
            out_vel[rg]   = vel;
            out_coord[rg] = coordv[rg] + upd_coord[rg] / degree[node] + vel;
        }
    }
}

extern "C" void kernel_launch(void* const* d_in, const int* in_sizes, int n_in,
                              void* d_out, int out_size, void* d_ws, size_t ws_size,
                              hipStream_t stream) {
    const float*     node_feat = (const float*)d_in[0];
    const float*     degree    = (const float*)d_in[1];
    const float*     coordv    = (const float*)d_in[2];
    const long long* ei        = (const long long*)d_in[3];
    const float*     velocity  = (const float*)d_in[4];
    const float* We1 = (const float*)d_in[5];  const float* be1 = (const float*)d_in[6];
    const float* We2 = (const float*)d_in[7];  const float* be2 = (const float*)d_in[8];
    const float* Wc1 = (const float*)d_in[9];  const float* bc1 = (const float*)d_in[10];
    const float* Wc2 = (const float*)d_in[11]; const float* bc2 = (const float*)d_in[12];
    const float* Wn1 = (const float*)d_in[13]; const float* bn1 = (const float*)d_in[14];
    const float* Wn2 = (const float*)d_in[15]; const float* bn2 = (const float*)d_in[16];
    const float* Wv1 = (const float*)d_in[17]; const float* bv1 = (const float*)d_in[18];
    const float* Wv2 = (const float*)d_in[19]; const float* bv2 = (const float*)d_in[20];

    // workspace layout
    float* upd_coord = (float*)d_ws;                       // 30000 f32 (reserve 32768)
    float* agg       = upd_coord + 32768;                  // 3,840,000 f32
    _Float16* hb     = (_Float16*)(agg + 3840000);
    _Float16* We1t = hb;                                    // 128*288
    _Float16* We2t = hb + 36864;                            // 128*128
    _Float16* Wc1t = hb + 53248;                            // 128*128
    _Float16* Wn1t = hb + 69632;                            // 128*256
    _Float16* Wn2t = hb + 102400;                           // 128*128
    _Float16* Wv1t = hb + 118784;                           // 128*128

    float* out       = (float*)d_out;
    float* out_coord = out;                                 // [N,3]
    float* out_node  = out + ROWS_N;                        // [N,3,128]
    float* out_vel   = out + ROWS_N + (size_t)ROWS_N * INDIM;

    // zero accumulators (upd_coord + agg, contiguous)
    hipMemsetAsync(d_ws, 0, (size_t)(32768 + 3840000) * sizeof(float), stream);

    egnn_prep_kernel<<<(128 * K1 + 255) / 256, 256, 0, stream>>>(
        We1, We2, Wc1, Wn1, Wn2, Wv1, We1t, We2t, Wc1t, Wn1t, Wn2t, Wv1t);

    egnn_edge_kernel<<<ROWS_E / 32 / 2, 64, 0, stream>>>(
        node_feat, coordv, ei, We1t, be1, We2t, be2, Wc1t, bc1, Wc2, bc2,
        agg, upd_coord);

    int ntiles = (ROWS_N + 15) / 16;
    egnn_node_kernel<<<(ntiles + 3) / 4, 128, 0, stream>>>(
        node_feat, degree, coordv, velocity,
        Wn1t, bn1, Wn2t, bn2, Wv1t, bv1, Wv2, bv2,
        agg, upd_coord, out_coord, out_node, out_vel);
}
